// GPTLanguageModel_46145128628411
// MI455X (gfx1250) — matrix operand order repaired and checked by reading, not compiled, not run on hardware
//
#include <hip/hip_runtime.h>
#include <cstdint>
#include <cstddef>

// ---------------------------------------------------------------------------
// GPT forward (B=64,T=256,C=384,H=6,L=6,V=252) for gfx1250 (MI455X).
// GEMMs: V_WMMA_F32_16X16X32_F16, 128x128 block tiles, TDM double-buffered
// LDS staging (tensor_load_to_lds + s_wait_tensorcnt).
// Attention: fused flash-style kernel, 8 WMMA per 32-key block.
// ---------------------------------------------------------------------------

typedef _Float16 h16 __attribute__((ext_vector_type(16)));
typedef _Float16 h8  __attribute__((ext_vector_type(8)));
typedef float    f8  __attribute__((ext_vector_type(8)));

#define NEMBD  384
#define NHEAD  6
#define HEADD  64
#define NLAYER 6
#define TSEQ   256
#define BBATCH 64
#define NTOK   (BBATCH*TSEQ)   /* 16384 tokens */
#define VOCABN 252
#define VPAD   256

#if __has_builtin(__builtin_amdgcn_tensor_load_to_lds) && \
    __has_builtin(__builtin_amdgcn_s_wait_tensorcnt)
#define USE_TDM 1
#else
#define USE_TDM 0
#endif

__device__ __forceinline__ f8 wmma16(h16 a, h16 b, f8 c) {
  // D = A(16x32 f16) * B(32x16 f16) + C(16x16 f32)
  return __builtin_amdgcn_wmma_f32_16x16x32_f16(false, a, false, b, (short)0, c,
                                                false, false);
}

#if USE_TDM
typedef unsigned int u32x4v __attribute__((ext_vector_type(4)));
typedef int          i32x8v __attribute__((ext_vector_type(8)));
typedef int          i32x4v __attribute__((ext_vector_type(4)));

// Issue one TDM descriptor: load a 128-row x 32-half 2D tile (element = 2B)
// from global (row stride = `stride` elements) into LDS at byte offset
// `ldsByteOff`, padding +8 DWORDs after every 16 DWORDs so the LDS image has
// the 48-half padded row stride the WMMA fragment readers expect.
__device__ __forceinline__ void tdm_load_tile(const _Float16* g,
                                              unsigned ldsByteOff,
                                              int width, int rows, int stride)
{
  unsigned long long ga = (unsigned long long)(uintptr_t)g;
  u32x4v g0;
  g0[0] = 1u;                                   // count=1 (valid user D#)
  g0[1] = ldsByteOff;                           // lds_addr (bytes)
  g0[2] = (unsigned)ga;                         // global_addr[31:0]
  g0[3] = (unsigned)((ga >> 32) & 0x01ffffffu)  // global_addr[56:32]
          | (2u << 30);                         // type=2 ("image")
  i32x8v g1;
  g1[0] = (int)((1u << 16)        // data_size = 1 -> 2 bytes
              | (1u << 20)        // pad_enable
              | (3u << 22)        // pad_interval: every 16 DWORDs
              | (7u << 25));      // pad_amount : 8 DWORDs (16 halves)
  g1[1] = (int)(((unsigned)width & 0xffffu) << 16);                 // dim0 lo
  g1[2] = (int)((((unsigned)width >> 16) & 0xffffu)                 // dim0 hi
              | (((unsigned)rows & 0xffffu) << 16));                // dim1 lo
  g1[3] = (int)((((unsigned)rows >> 16) & 0xffffu)                  // dim1 hi
              | (32u << 16));                                       // tile_dim0
  g1[4] = 128;                                  // tile_dim1=128, tile_dim2=0
  g1[5] = stride;                               // tensor_dim0_stride[31:0]
  g1[6] = 0;
  g1[7] = 0;
  i32x4v z4 = {0, 0, 0, 0};                     // groups 2/3 unused (2D)
  i32x8v z8 = {0, 0, 0, 0, 0, 0, 0, 0};
  __builtin_amdgcn_tensor_load_to_lds(g0, g1, z4, z4, z8, 0);
}
#endif

// ---------------------------------------------------------------------------
// Weight convert: f32 (K x N) row-major -> f16 transposed (NP x K),
// rows n >= N zero-padded (pads lm_W 252 -> 256). Grid covers NP*K exactly.
// ---------------------------------------------------------------------------
__global__ __launch_bounds__(256) void convt_kernel(
    const float* __restrict__ src, _Float16* __restrict__ dst,
    int K, int N, int NP)
{
  (void)NP;
  int i = blockIdx.x * 256 + threadIdx.x;
  int n = i / K;
  int k = i - n * K;
  float v = (n < N) ? src[(size_t)k * N + n] : 0.0f;
  dst[i] = (_Float16)v;
}

// ---------------------------------------------------------------------------
__global__ __launch_bounds__(256) void embed_kernel(
    const int* __restrict__ idx, const float* __restrict__ tok,
    const float* __restrict__ pos, float* __restrict__ x)
{
  int i = blockIdx.x * 256 + threadIdx.x;        // NTOK*NEMBD exact
  int trow = i / NEMBD;
  int c = i - trow * NEMBD;
  int tpos = trow & (TSEQ - 1);
  x[i] = tok[(size_t)idx[trow] * NEMBD + c] + pos[(size_t)tpos * NEMBD + c];
}

// ---------------------------------------------------------------------------
// LayerNorm: one wave per token row (C=384 -> 12 f32 per lane), f16 output.
// ---------------------------------------------------------------------------
__global__ __launch_bounds__(256) void ln_kernel(
    const float* __restrict__ x, const float* __restrict__ g,
    const float* __restrict__ bta, _Float16* __restrict__ out)
{
  int gw   = (blockIdx.x * 256 + threadIdx.x) >> 5;
  int lane = threadIdx.x & 31;
  const float* xr = x + (size_t)gw * NEMBD;
  float v[12];
  float s = 0.0f;
#pragma unroll
  for (int i = 0; i < 12; ++i) { v[i] = xr[lane + 32 * i]; s += v[i]; }
#pragma unroll
  for (int m = 1; m < 32; m <<= 1) s += __shfl_xor(s, m, 32);
  float mu = s * (1.0f / NEMBD);
  float var = 0.0f;
#pragma unroll
  for (int i = 0; i < 12; ++i) { float d = v[i] - mu; var += d * d; }
#pragma unroll
  for (int m = 1; m < 32; m <<= 1) var += __shfl_xor(var, m, 32);
  float rstd = rsqrtf(var * (1.0f / NEMBD) + 1e-5f);
#pragma unroll
  for (int i = 0; i < 12; ++i) {
    int c = lane + 32 * i;
    out[(size_t)gw * NEMBD + c] = (_Float16)((v[i] - mu) * rstd * g[c] + bta[c]);
  }
}

// ---------------------------------------------------------------------------
// Tiled WMMA GEMM: C[M,N] = A[M,K] (f16 row-major) * BT[N,K] (f16, W^T).
// Block tile 128(M) x 128(N), K-step 32, 8 waves each owning 32x64
// (2x4 WMMA tiles -> 8 v_wmma per K-step, A-frags reused across 4 B-frags).
// K-tiles staged to LDS by the Tensor Data Mover, double-buffered.
// MODE 0: f16 out, no bias (q/k/v)
// MODE 1: f16 out, bias + relu (MLP up)
// MODE 2: f32 residual += acc + bias (attn proj, MLP down)
// MODE 3: f32 out, bias, columns masked to ldOut (lm_head, N padded to 256)
// ---------------------------------------------------------------------------
template <int MODE>
__global__ __launch_bounds__(256) void gemm_wmma(
    const _Float16* __restrict__ A, const _Float16* __restrict__ BT,
    const float* __restrict__ bias,
    _Float16* __restrict__ outH, float* __restrict__ outF,
    int K, int N, int ldOut)
{
  __shared__ _Float16 As[2][128 * 48];   // 128 rows x 32 halves (+16 pad)
  __shared__ _Float16 Bs[2][128 * 48];

  const int t    = threadIdx.x;
  const int lane = t & 31;
  const int wv   = t >> 5;
  const int wM   = wv >> 1;            // 0..3  (32-row slice)
  const int wN   = wv & 1;             // 0..1  (64-col slice)
  const int hl   = lane >> 4;          // lane half (ISA fragment layout)
  const int ml   = lane & 15;
  const int Mb   = blockIdx.x * 128;
  const int Nb   = blockIdx.y * 128;

  f8 acc[2][4];
#pragma unroll
  for (int i = 0; i < 2; ++i)
#pragma unroll
    for (int j = 0; j < 4; ++j)
#pragma unroll
      for (int e = 0; e < 8; ++e) acc[i][j][e] = 0.0f;

  const _Float16* Abase = A  + (size_t)Mb * K;
  const _Float16* Bbase = BT + (size_t)Nb * K;

#if USE_TDM
  const unsigned offA[2] = { (unsigned)(uintptr_t)(void*)&As[0][0],
                             (unsigned)(uintptr_t)(void*)&As[1][0] };
  const unsigned offB[2] = { (unsigned)(uintptr_t)(void*)&Bs[0][0],
                             (unsigned)(uintptr_t)(void*)&Bs[1][0] };
  if (wv == 0) {
    tdm_load_tile(Abase, offA[0], K, NTOK - Mb, K);
    tdm_load_tile(Bbase, offB[0], K, N - Nb, K);
  }
#endif

  int buf = 0;
  for (int k0 = 0; k0 < K; k0 += 32) {
#if USE_TDM
    if (wv == 0) {
      if (k0 + 32 < K) {
        // prefetch next K-tile into the other buffer, then wait for current
        tdm_load_tile(Abase + k0 + 32, offA[buf ^ 1], K - k0 - 32, NTOK - Mb, K);
        tdm_load_tile(Bbase + k0 + 32, offB[buf ^ 1], K - k0 - 32, N - Nb, K);
        __builtin_amdgcn_s_wait_tensorcnt((short)2);
      } else {
        __builtin_amdgcn_s_wait_tensorcnt((short)0);
      }
    }
    __syncthreads();                       // current buffer visible to all
#else
    __syncthreads();                       // prior-iteration reads done
    {
      int row = t >> 1;
      int col = (t & 1) * 16;
      *(h16*)&As[buf][row * 48 + col] =
          *(const h16*)(Abase + (size_t)row * K + k0 + col);
      *(h16*)&Bs[buf][row * 48 + col] =
          *(const h16*)(Bbase + (size_t)row * K + k0 + col);
      if (k0 + 32 < K) {
        __builtin_prefetch(Abase + (size_t)row * K + k0 + 32 + col, 0, 1);
        __builtin_prefetch(Bbase + (size_t)row * K + k0 + 32 + col, 0, 1);
      }
    }
    __syncthreads();
#endif

    h16 af[2], bf[4];
#pragma unroll
    for (int tm = 0; tm < 2; ++tm) {
      int mr = wM * 32 + tm * 16 + ml;
      af[tm] = *(const h16*)&As[buf][mr * 48 + hl * 16];    // K = hl*16 + e
    }
#pragma unroll
    for (int tn = 0; tn < 4; ++tn) {
      int nr = wN * 64 + tn * 16 + ml;
      bf[tn] = *(const h16*)&Bs[buf][nr * 48 + hl * 16];    // K = hl*16 + e
    }
#pragma unroll
    for (int tm = 0; tm < 2; ++tm)
#pragma unroll
      for (int tn = 0; tn < 4; ++tn)
        acc[tm][tn] = wmma16(af[tm], bf[tn], acc[tm][tn]);

    __syncthreads();                       // all reads done before re-staging
    buf ^= 1;
  }

#pragma unroll
  for (int tm = 0; tm < 2; ++tm) {
#pragma unroll
    for (int tn = 0; tn < 4; ++tn) {
      int col = Nb + wN * 64 + tn * 16 + ml;
      float bia = 0.0f;
      if (MODE != 0) bia = (MODE == 3 && col >= ldOut) ? 0.0f : bias[col];
#pragma unroll
      for (int r = 0; r < 8; ++r) {
        int row = Mb + wM * 32 + tm * 16 + r + 8 * hl;  // C layout: M=r+8*half
        float v = acc[tm][tn][r] + bia;
        if (MODE == 0)
          outH[(size_t)row * ldOut + col] = (_Float16)acc[tm][tn][r];
        else if (MODE == 1)
          outH[(size_t)row * ldOut + col] = (_Float16)fmaxf(v, 0.0f);
        else if (MODE == 2)
          outF[(size_t)row * ldOut + col] += v;
        else if (col < ldOut)
          outF[(size_t)row * ldOut + col] = v;
      }
    }
  }
}

// ---------------------------------------------------------------------------
// Fused causal attention (flash-style, online softmax).
// Block = one (batch, head, 128-query slab); 8 waves x 16 queries each.
// Per 32-key block: 4 WMMA for S = Q*K^T (K=64), 4 WMMA for O += P*V.
// ---------------------------------------------------------------------------
__global__ __launch_bounds__(256) void attn_kernel(
    const _Float16* __restrict__ qf, const _Float16* __restrict__ kf,
    const _Float16* __restrict__ vf, _Float16* __restrict__ of)
{
  __shared__ _Float16 VT[64 * 48];       // V block transposed: [d][key]
  __shared__ _Float16 Pb[8 * 16 * 40];   // per-wave P staging [16][32]+pad

  const int t    = threadIdx.x;
  const int lane = t & 31;
  const int wv   = t >> 5;
  const int hl   = lane >> 4;
  const int ml   = lane & 15;
  const int bh   = blockIdx.x >> 1;
  const int qh   = blockIdx.x & 1;
  const int bb   = bh / NHEAD;
  const int hh   = bh % NHEAD;
  const size_t tokbase = (size_t)bb * TSEQ;
  const int q0 = qh * 128 + wv * 16;

  // Q fragments (16 queries x 64 dims = two 16x32 A-fragments), loaded once.
  h16 qa[2];
#pragma unroll
  for (int kc = 0; kc < 2; ++kc) {
    const _Float16* qp = qf + (tokbase + q0 + ml) * NEMBD + hh * HEADD + kc * 32;
    qa[kc] = *(const h16*)(qp + hl * 16);
  }

  f8 o[4];
#pragma unroll
  for (int d = 0; d < 4; ++d)
#pragma unroll
    for (int e = 0; e < 8; ++e) o[d][e] = 0.0f;
  float mrow[8], lrow[8];
#pragma unroll
  for (int r = 0; r < 8; ++r) { mrow[r] = -__builtin_inff(); lrow[r] = 0.0f; }

  _Float16* Pw = &Pb[wv * 640];
  const int nkb = 4 + qh * 4;            // causal: uniform trips per block

  for (int kb = 0; kb < nkb; ++kb) {
    __syncthreads();
    { // cooperative stage of V block (32 keys x 64 dims), transposed in LDS
      int key = t >> 3;
      int dg  = (t & 7) * 8;
      h8 vv = *(const h8*)(vf + (tokbase + kb * 32 + key) * NEMBD +
                           hh * HEADD + dg);
#pragma unroll
      for (int e = 0; e < 8; ++e) VT[(dg + e) * 48 + key] = vv[e];
    }
    __syncthreads();

    // S (16 queries x 32 keys) = Q * K^T
    f8 s[2];
#pragma unroll
    for (int tn = 0; tn < 2; ++tn) {
#pragma unroll
      for (int e = 0; e < 8; ++e) s[tn][e] = 0.0f;
#pragma unroll
      for (int kc = 0; kc < 2; ++kc) {
        h16 kfr = *(const h16*)(kf + (tokbase + kb * 32 + tn * 16 + ml) * NEMBD +
                                hh * HEADD + kc * 32 + hl * 16);
        s[tn] = wmma16(qa[kc], kfr, s[tn]);
      }
    }

    // scale + causal mask (branchless; EXEC stays all-ones)
#pragma unroll
    for (int r = 0; r < 8; ++r) {
      int qg = q0 + r + 8 * hl;
#pragma unroll
      for (int tn = 0; tn < 2; ++tn) {
        int kg = kb * 32 + tn * 16 + ml;
        float v = s[tn][r] * 0.125f;               // HEAD^-0.5
        s[tn][r] = (kg <= qg) ? v : -1.0e30f;
      }
    }

    // online softmax update (row reductions over 16-lane half)
#pragma unroll
    for (int r = 0; r < 8; ++r) {
      float sm = fmaxf(s[0][r], s[1][r]);
#pragma unroll
      for (int m = 1; m < 16; m <<= 1) sm = fmaxf(sm, __shfl_xor(sm, m, 32));
      float mnew = fmaxf(mrow[r], sm);
      float corr = __expf(mrow[r] - mnew);
      float p0 = __expf(s[0][r] - mnew);
      float p1 = __expf(s[1][r] - mnew);
      s[0][r] = p0; s[1][r] = p1;
      float rs = p0 + p1;
#pragma unroll
      for (int m = 1; m < 16; m <<= 1) rs += __shfl_xor(rs, m, 32);
      lrow[r] = lrow[r] * corr + rs;
      mrow[r] = mnew;
#pragma unroll
      for (int d = 0; d < 4; ++d) o[d][r] *= corr;
    }

    // P: C-layout -> LDS -> A-fragment reshape
#pragma unroll
    for (int r = 0; r < 8; ++r) {
      int prow = r + 8 * hl;
      Pw[prow * 40 + ml]      = (_Float16)s[0][r];
      Pw[prow * 40 + 16 + ml] = (_Float16)s[1][r];
    }
    __syncthreads();
    h16 pa;
    {
      h8 lo = *(const h8*)&Pw[ml * 40 + hl * 8];
      h8 hi = *(const h8*)&Pw[ml * 40 + 16 + hl * 8];
#pragma unroll
      for (int e = 0; e < 8; ++e) { pa[e] = lo[e]; pa[e + 8] = hi[e]; }
    }
    // O += P * V
#pragma unroll
    for (int d = 0; d < 4; ++d) {
      h8 vlo = *(const h8*)&VT[(d * 16 + ml) * 48 + hl * 8];
      h8 vhi = *(const h8*)&VT[(d * 16 + ml) * 48 + 16 + hl * 8];
      h16 vb;
#pragma unroll
      for (int e = 0; e < 8; ++e) { vb[e] = vlo[e]; vb[e + 8] = vhi[e]; }
      o[d] = wmma16(pa, vb, o[d]);
    }
  }

#pragma unroll
  for (int r = 0; r < 8; ++r) {
    float inv = 1.0f / lrow[r];
    size_t tok = tokbase + q0 + r + 8 * hl;
#pragma unroll
    for (int d = 0; d < 4; ++d)
      of[tok * NEMBD + hh * HEADD + d * 16 + ml] = (_Float16)(o[d][r] * inv);
  }
}

// ---------------------------------------------------------------------------
// Loss: one wave per token row of logits (252 cols), atomic mean NLL.
// ---------------------------------------------------------------------------
__global__ __launch_bounds__(256) void loss_kernel(
    const float* __restrict__ logits, const int* __restrict__ tgt,
    float* __restrict__ loss)
{
  int gw   = (blockIdx.x * 256 + threadIdx.x) >> 5;
  int lane = threadIdx.x & 31;
  const float* lr = logits + (size_t)gw * VOCABN;
  float vals[8];
  float mx = -__builtin_inff();
#pragma unroll
  for (int i = 0; i < 8; ++i) {
    int c = lane + 32 * i;
    vals[i] = (c < VOCABN) ? lr[c] : -__builtin_inff();
    mx = fmaxf(mx, vals[i]);
  }
#pragma unroll
  for (int m = 1; m < 32; m <<= 1) mx = fmaxf(mx, __shfl_xor(mx, m, 32));
  float s = 0.0f;
#pragma unroll
  for (int i = 0; i < 8; ++i) s += __expf(vals[i] - mx);
#pragma unroll
  for (int m = 1; m < 32; m <<= 1) s += __shfl_xor(s, m, 32);
  if (lane == 0) {
    float lp = lr[tgt[gw]] - mx - __logf(s);
    atomicAdd(loss, -lp * (1.0f / NTOK));
  }
}

__global__ void zero_kernel(float* p) { *p = 0.0f; }

// ---------------------------------------------------------------------------
extern "C" void kernel_launch(void* const* d_in, const int* in_sizes, int n_in,
                              void* d_out, int out_size, void* d_ws,
                              size_t ws_size, hipStream_t stream)
{
  (void)in_sizes; (void)n_in; (void)out_size; (void)ws_size;

  const int*   idx     = (const int*)  d_in[0];
  const int*   targets = (const int*)  d_in[1];
  const float* tok_emb = (const float*)d_in[2];
  const float* pos_emb = (const float*)d_in[3];
  const float* Wq      = (const float*)d_in[4];
  const float* Wk      = (const float*)d_in[5];
  const float* Wv      = (const float*)d_in[6];
  const float* Wo      = (const float*)d_in[7];
  const float* bo      = (const float*)d_in[8];
  const float* W1      = (const float*)d_in[9];
  const float* b1      = (const float*)d_in[10];
  const float* W2      = (const float*)d_in[11];
  const float* b2      = (const float*)d_in[12];
  const float* ln1_g   = (const float*)d_in[13];
  const float* ln1_b   = (const float*)d_in[14];
  const float* ln2_g   = (const float*)d_in[15];
  const float* ln2_b   = (const float*)d_in[16];
  const float* lnf_g   = (const float*)d_in[17];
  const float* lnf_b   = (const float*)d_in[18];
  const float* lm_W    = (const float*)d_in[19];
  const float* lm_b    = (const float*)d_in[20];

  float* logits = (float*)d_out;                       // [NTOK, 252]
  float* loss   = logits + (size_t)NTOK * VOCABN;      // scalar

  // ---- workspace arena (all offsets 256B aligned) ----
  size_t off = 0;
  auto arena = [&](size_t bytes) -> char* {
    char* p = (char*)d_ws + off;
    off += (bytes + 255) & ~(size_t)255;
    return p;
  };
  const size_t SZ_ATT = (size_t)NLAYER * NEMBD * NEMBD * 2;       // 1.77MB each
  const size_t SZ_MLP = (size_t)NLAYER * NEMBD * 4 * NEMBD * 2;   // 7.08MB
  const size_t SZ_H   = (size_t)NTOK * NEMBD * 2;                 // 12.6MB
  _Float16* wtq  = (_Float16*)arena(SZ_ATT);
  _Float16* wtk  = (_Float16*)arena(SZ_ATT);
  _Float16* wtv  = (_Float16*)arena(SZ_ATT);
  _Float16* wto  = (_Float16*)arena(SZ_ATT);
  _Float16* wt1  = (_Float16*)arena(SZ_MLP);
  _Float16* wt2  = (_Float16*)arena(SZ_MLP);
  _Float16* wtlm = (_Float16*)arena((size_t)VPAD * NEMBD * 2);
  float*    x    = (float*)   arena((size_t)NTOK * NEMBD * 4);
  _Float16* hbuf = (_Float16*)arena(SZ_H);
  _Float16* big  = (_Float16*)arena((size_t)NTOK * 4 * NEMBD * 2); // 50.3MB
  _Float16* qbuf = big;                         // aliases: q/k/v/attn share
  _Float16* kbuf = big + (size_t)NTOK * NEMBD;  // 'big' with the MLP hidden
  _Float16* vbuf = big + (size_t)NTOK * NEMBD * 2;
  _Float16* abuf = big + (size_t)NTOK * NEMBD * 3;
  _Float16* mbuf = big;                         // MLP hidden 16384x1536

  // ---- weight conversion (f32 -> f16 transposed) ----
  for (int l = 0; l < NLAYER; ++l) {
    size_t oA = (size_t)l * NEMBD * NEMBD;
    size_t oM = (size_t)l * NEMBD * 4 * NEMBD;
    convt_kernel<<<576, 256, 0, stream>>>(Wq + oA, wtq + oA, NEMBD, NEMBD, NEMBD);
    convt_kernel<<<576, 256, 0, stream>>>(Wk + oA, wtk + oA, NEMBD, NEMBD, NEMBD);
    convt_kernel<<<576, 256, 0, stream>>>(Wv + oA, wtv + oA, NEMBD, NEMBD, NEMBD);
    convt_kernel<<<576, 256, 0, stream>>>(Wo + oA, wto + oA, NEMBD, NEMBD, NEMBD);
    convt_kernel<<<2304, 256, 0, stream>>>(W1 + oM, wt1 + oM, NEMBD, 4 * NEMBD, 4 * NEMBD);
    convt_kernel<<<2304, 256, 0, stream>>>(W2 + oM, wt2 + oM, 4 * NEMBD, NEMBD, NEMBD);
  }
  convt_kernel<<<384, 256, 0, stream>>>(lm_W, wtlm, NEMBD, VOCABN, VPAD);

  // ---- embedding ----
  embed_kernel<<<(NTOK * NEMBD) / 256, 256, 0, stream>>>(idx, tok_emb, pos_emb, x);

  const dim3 g384(NTOK / 128, 3);    // N=384
  const dim3 g1536(NTOK / 128, 12);  // N=1536
  const dim3 g256(NTOK / 128, 2);    // N=256 (padded lm head)
  const int  lnBlocks = NTOK / 8;    // 8 waves per 256-thread block

  for (int l = 0; l < NLAYER; ++l) {
    size_t oA = (size_t)l * NEMBD * NEMBD;
    size_t oM = (size_t)l * NEMBD * 4 * NEMBD;
    ln_kernel<<<lnBlocks, 256, 0, stream>>>(x, ln1_g + l * NEMBD, ln1_b + l * NEMBD, hbuf);
    gemm_wmma<0><<<g384, 256, 0, stream>>>(hbuf, wtq + oA, nullptr, qbuf, nullptr, NEMBD, NEMBD, NEMBD);
    gemm_wmma<0><<<g384, 256, 0, stream>>>(hbuf, wtk + oA, nullptr, kbuf, nullptr, NEMBD, NEMBD, NEMBD);
    gemm_wmma<0><<<g384, 256, 0, stream>>>(hbuf, wtv + oA, nullptr, vbuf, nullptr, NEMBD, NEMBD, NEMBD);
    attn_kernel<<<BBATCH * NHEAD * 2, 256, 0, stream>>>(qbuf, kbuf, vbuf, abuf);
    gemm_wmma<2><<<g384, 256, 0, stream>>>(abuf, wto + oA, bo + l * NEMBD, nullptr, x, NEMBD, NEMBD, NEMBD);
    ln_kernel<<<lnBlocks, 256, 0, stream>>>(x, ln2_g + l * NEMBD, ln2_b + l * NEMBD, hbuf);
    gemm_wmma<1><<<g1536, 256, 0, stream>>>(hbuf, wt1 + oM, b1 + l * 4 * NEMBD, mbuf, nullptr, NEMBD, 4 * NEMBD, 4 * NEMBD);
    gemm_wmma<2><<<g384, 256, 0, stream>>>(mbuf, wt2 + oM, b2 + l * NEMBD, nullptr, x, 4 * NEMBD, NEMBD, NEMBD);
  }

  // ---- final LN + lm_head + loss ----
  ln_kernel<<<lnBlocks, 256, 0, stream>>>(x, lnf_g, lnf_b, hbuf);
  gemm_wmma<3><<<g256, 256, 0, stream>>>(hbuf, wtlm, lm_b, nullptr, logits, NEMBD, VPAD, VOCABN);
  zero_kernel<<<1, 1, 0, stream>>>(loss);
  loss_kernel<<<NTOK / 8, 256, 0, stream>>>(logits, targets, loss);
}
